// AFM_13572096655929
// MI455X (gfx1250) — compile-verified, matrix operand order checked by reading, not run on hardware
//
#include <hip/hip_runtime.h>

typedef __attribute__((ext_vector_type(16))) __bf16 v16bf;
typedef __attribute__((ext_vector_type(8)))  float  v8f;

#define WMMA_BF16(a, b, c) \
  __builtin_amdgcn_wmma_f32_16x16x32_bf16(false, (a), false, (b), (short)0, (c), false, false)

static constexpr int kB   = 4096;
static constexpr int kF   = 26;
static constexpr int kV   = 100000;
static constexpr int kD   = 64;
static constexpr int kND  = 13;
static constexpr int kNP  = 325;   // F*(F-1)/2
static constexpr int kNPT = 21;    // ceil(325/16) row tiles
static constexpr int kXS  = 80;    // padded row stride for x in workspace

// Packed bf16 weight fragment layout:
//   frag (kt, nt) at element offset ((kt*NT + nt)*32 + lane)*16, 16 bf16/lane.
//   lane: col = lane&15, klo = (lane>>4)*8;
//   elem i<8 -> k = kt*32+klo+i ; elem i>=8 -> k = kt*32+klo+16+(i-8)
static constexpr int kPWa = 2 * 4 * 32 * 16;    //  4096 bf16 (64x64)
static constexpr int kPW1 = 3 * 16 * 32 * 16;   // 24576 bf16 (96x256, K padded)
static constexpr int kPW2 = 8 * 8 * 32 * 16;    // 32768 bf16 (256x128)
static constexpr int kPW3 = 4 * 4 * 32 * 16;    //  8192 bf16 (128x64)

// ---------------------------------------------------------------------------
// Weight pre-pack: f32 row-major (Kreal x N) -> bf16 fragment-native layout,
// zero-padded to KT*32 in K. One thread per lane-slice (16 elements).
// ---------------------------------------------------------------------------
__global__ __launch_bounds__(256) void afm_pack(
    const float* __restrict__ src, unsigned short* __restrict__ dst,
    int Kreal, int KT, int N)
{
  int t = blockIdx.x * blockDim.x + threadIdx.x;
  int NT = N >> 4;
  int total = KT * NT * 32;
  if (t >= total) return;
  int lane = t & 31;
  int fi   = t >> 5;
  int kt   = fi / NT, nt = fi % NT;
  int colv = lane & 15;
  int klo  = (lane >> 4) << 3;
  unsigned short* d = dst + (size_t)t * 16;
  #pragma unroll
  for (int i = 0; i < 8; ++i) {
    int k0 = kt * 32 + klo + i;
    int k1 = k0 + 16;
    __bf16 h0 = (__bf16)((k0 < Kreal) ? src[k0 * N + nt * 16 + colv] : 0.0f);
    __bf16 h1 = (__bf16)((k1 < Kreal) ? src[k1 * N + nt * 16 + colv] : 0.0f);
    d[i]     = *(unsigned short*)&h0;
    d[i + 8] = *(unsigned short*)&h1;
  }
}

__device__ __forceinline__ v16bf load_frag(const unsigned short* pw,
                                           int kt, int NT, int nt, int lane) {
  return *(const v16bf*)(pw + ((size_t)(kt * NT + nt) * 32 + lane) * 16);
}

// ---------------------------------------------------------------------------
// Kernel 1: embedding gather + pairwise products + attention (WMMA) + softmax
// + attention pooling. One workgroup per batch sample, 256 threads = 8 waves.
// ---------------------------------------------------------------------------
__global__ __launch_bounds__(256) void afm_attention(
    const int* __restrict__ sx, const float* __restrict__ dx,
    const float* __restrict__ emb, const unsigned short* __restrict__ pWa,
    const float* __restrict__ ba, const float* __restrict__ dW,
    const float* __restrict__ db, float* __restrict__ X)
{
  __shared__ float s_emb[kF * kD];          // gathered embeddings (f32)
  __shared__ int   s_idx[kF];
  __shared__ unsigned char s_p1[kNP], s_p2[kNP];
  __shared__ float s_sc[kNPT * 16];         // scores -> exp weights
  __shared__ float s_red[256];
  __shared__ float s_part[4][kD];

  const int b    = blockIdx.x;
  const int tid  = threadIdx.x;
  const int lane = tid & 31;
  const int wave = tid >> 5;
  const int col  = lane & 15;               // N index within tile
  const int klo  = (lane >> 4) << 3;        // 0 or 8: K sub-offset per layout

  if (tid < kF) s_idx[tid] = sx[b * kF + tid];
  // pair index tables (np.triu_indices order)
  for (int p = tid; p < kNP; p += 256) {
    int i = 0, rem = p;
    while (rem >= kF - 1 - i) { rem -= kF - 1 - i; ++i; }
    s_p1[p] = (unsigned char)i;
    s_p2[p] = (unsigned char)(i + 1 + rem);
  }
  __syncthreads();

  // gather embeddings: 26 rows of 64 floats
  for (int t = tid; t < kF * kD; t += 256) {
    int f = t >> 6, d = t & 63;
    long idx = ((long)f * kV + s_idx[f]) * (long)kD + d;
    s_emb[t] = emb[idx];
  }

  // B fragments of att_w_W from packed bf16 (2 b128 loads each)
  v16bf Bfrag[2][4];
  float biasv[4], dvn[4];
  #pragma unroll
  for (int kt = 0; kt < 2; ++kt)
    #pragma unroll
    for (int nt = 0; nt < 4; ++nt)
      Bfrag[kt][nt] = load_frag(pWa, kt, 4, nt, lane);
  #pragma unroll
  for (int nt = 0; nt < 4; ++nt) {
    biasv[nt] = ba[nt * 16 + col];
    dvn[nt]   = dW[nt * 16 + col];
  }
  const float dbias = db[0];
  __syncthreads();

  // attention matmul over pair-row tiles
  for (int rt = wave; rt < kNPT; rt += 8) {
    int row  = rt * 16 + col;                // pair index for this lane's A row
    int rowc = row < kNP ? row : kNP - 1;    // clamp: padded rows are ignored
    const float4* E1 = (const float4*)(s_emb + (int)s_p1[rowc] * kD);
    const float4* E2 = (const float4*)(s_emb + (int)s_p2[rowc] * kD);
    v16bf Af[2];
    #pragma unroll
    for (int kt = 0; kt < 2; ++kt) {
      int q = (kt * 32 + klo) >> 2;          // float4 index of first K group
      float ea[16], eb[16];
      *(float4*)(ea + 0)  = E1[q];
      *(float4*)(ea + 4)  = E1[q + 1];
      *(float4*)(ea + 8)  = E1[q + 4];
      *(float4*)(ea + 12) = E1[q + 5];
      *(float4*)(eb + 0)  = E2[q];
      *(float4*)(eb + 4)  = E2[q + 1];
      *(float4*)(eb + 8)  = E2[q + 4];
      *(float4*)(eb + 12) = E2[q + 5];
      v16bf va;
      #pragma unroll
      for (int i = 0; i < 16; ++i) va[i] = (__bf16)(ea[i] * eb[i]);
      Af[kt] = va;
    }
    float ssum[8];
    #pragma unroll
    for (int v = 0; v < 8; ++v) ssum[v] = 0.0f;
    #pragma unroll
    for (int nt = 0; nt < 4; ++nt) {
      v8f acc = {};
      acc = WMMA_BF16(Af[0], Bfrag[0][nt], acc);
      acc = WMMA_BF16(Af[1], Bfrag[1][nt], acc);
      #pragma unroll
      for (int v = 0; v < 8; ++v) {
        float a = fmaxf(acc[v] + biasv[nt], 0.0f);   // ReLU
        ssum[v] += a * dvn[nt];                      // fold in att_d_W
      }
    }
    // reduce over the 16 N-lanes within each half-wave; write scores
    #pragma unroll
    for (int v = 0; v < 8; ++v) {
      float s = ssum[v];
      s += __shfl_xor(s, 1, 32);
      s += __shfl_xor(s, 2, 32);
      s += __shfl_xor(s, 4, 32);
      s += __shfl_xor(s, 8, 32);
      int m = rt * 16 + v + ((lane >> 4) << 3);
      if ((lane & 15) == 0 && m < kNP) s_sc[m] = s + dbias;
    }
  }
  __syncthreads();

  // softmax over 325 scores
  float lm = -1e30f;
  for (int p = tid; p < kNP; p += 256) lm = fmaxf(lm, s_sc[p]);
  s_red[tid] = lm;
  __syncthreads();
  for (int k = 128; k > 0; k >>= 1) {
    if (tid < k) s_red[tid] = fmaxf(s_red[tid], s_red[tid + k]);
    __syncthreads();
  }
  float mx = s_red[0];
  __syncthreads();
  float ls = 0.0f;
  for (int p = tid; p < kNP; p += 256) {
    float e = __expf(s_sc[p] - mx);
    s_sc[p] = e;
    ls += e;
  }
  s_red[tid] = ls;
  __syncthreads();
  for (int k = 128; k > 0; k >>= 1) {
    if (tid < k) s_red[tid] += s_red[tid + k];
    __syncthreads();
  }
  float inv = 1.0f / s_red[0];

  // attention pooling: att_out[d] = inv * sum_p e_p * e1[d]*e2[d]
  int d = tid & 63, g = tid >> 6;
  float accd = 0.0f;
  for (int p = g; p < kNP; p += 4)
    accd += s_sc[p] * s_emb[(int)s_p1[p] * kD + d] * s_emb[(int)s_p2[p] * kD + d];
  s_part[g][d] = accd;
  __syncthreads();
  if (tid < 64) {
    float o = (s_part[0][tid] + s_part[1][tid] + s_part[2][tid] + s_part[3][tid]) * inv;
    X[b * kXS + tid] = o;
  } else if (tid < 77) {
    X[b * kXS + tid] = dx[b * kND + (tid - 64)];
  }
}

// ---------------------------------------------------------------------------
// Kernel 2: batchnorm statistics (mean/var over B) -> fused scale/shift
// ---------------------------------------------------------------------------
__global__ __launch_bounds__(256) void afm_bnstats(
    const float* __restrict__ X, const float* __restrict__ gamma,
    const float* __restrict__ beta, float* __restrict__ scale,
    float* __restrict__ shift)
{
  const int f = blockIdx.x;
  const int tid = threadIdx.x;
  __shared__ float r1[256], r2[256];
  float s = 0.0f, s2 = 0.0f;
  for (int b = tid; b < kB; b += 256) {
    float v = X[b * kXS + f];
    s += v;
    s2 += v * v;
  }
  r1[tid] = s; r2[tid] = s2;
  __syncthreads();
  for (int k = 128; k > 0; k >>= 1) {
    if (tid < k) { r1[tid] += r1[tid + k]; r2[tid] += r2[tid + k]; }
    __syncthreads();
  }
  if (tid == 0) {
    float mu  = r1[0] * (1.0f / kB);
    float var = r2[0] * (1.0f / kB) - mu * mu;
    float sc  = gamma[f] * rsqrtf(var + 1e-5f);
    scale[f] = sc;
    shift[f] = beta[f] - mu * sc;
  }
}

// ---------------------------------------------------------------------------
// Kernel 3: MLP 77->256->128->64->1 via bf16 WMMA, 16 batch rows per block.
// 128 threads = 4 waves. Weights come pre-packed bf16 from workspace.
// ---------------------------------------------------------------------------
__device__ __forceinline__ v16bf afrag_from_lds(const float* base, int stride4,
                                                int col, int kt, int klo) {
  // base indexed as float4; row = col, k group = kt*32+klo (+16)
  const float4* H = (const float4*)base + (size_t)col * stride4;
  int q = (kt * 32 + klo) >> 2;
  float ha[16];
  *(float4*)(ha + 0)  = H[q];
  *(float4*)(ha + 4)  = H[q + 1];
  *(float4*)(ha + 8)  = H[q + 4];
  *(float4*)(ha + 12) = H[q + 5];
  v16bf va;
  #pragma unroll
  for (int i = 0; i < 16; ++i) va[i] = (__bf16)ha[i];
  return va;
}

__global__ __launch_bounds__(128) void afm_mlp(
    const float* __restrict__ X, const float* __restrict__ scale,
    const float* __restrict__ shift,
    const unsigned short* __restrict__ pW1, const float* __restrict__ b1,
    const unsigned short* __restrict__ pW2, const float* __restrict__ b2,
    const unsigned short* __restrict__ pW3, const float* __restrict__ b3,
    const float* __restrict__ wf, const float* __restrict__ bfs,
    float* __restrict__ out)
{
  __shared__ float h0[16 * 96];
  __shared__ float h1[16 * 256];
  __shared__ float h2[16 * 128];
  __shared__ float h3[16 * 64];

  const int b0   = blockIdx.x * 16;
  const int tid  = threadIdx.x;
  const int lane = tid & 31;
  const int wave = tid >> 5;
  const int col  = lane & 15;
  const int klo  = (lane >> 4) << 3;
  const int mrow = ((lane >> 4) << 3);   // C/D row base for this half-wave

  // load + normalize input, pad 77 -> 96 with zeros
  for (int t = tid; t < 16 * 96; t += 128) {
    int r = t / 96, c = t % 96;
    float v = 0.0f;
    if (c < 77) v = X[(b0 + r) * kXS + c] * scale[c] + shift[c];
    h0[t] = v;
  }
  __syncthreads();

  // ---- layer 1: (16x96) @ (96x256) ----
  v16bf A1[3];
  #pragma unroll
  for (int kt = 0; kt < 3; ++kt) A1[kt] = afrag_from_lds(h0, 96 / 4, col, kt, klo);
  for (int nt = wave; nt < 16; nt += 4) {
    v8f acc = {};
    #pragma unroll
    for (int kt = 0; kt < 3; ++kt)
      acc = WMMA_BF16(A1[kt], load_frag(pW1, kt, 16, nt, lane), acc);
    float bias = b1[nt * 16 + col];
    #pragma unroll
    for (int v = 0; v < 8; ++v)
      h1[(v + mrow) * 256 + nt * 16 + col] = fmaxf(acc[v] + bias, 0.0f);
  }
  __syncthreads();

  // ---- layer 2: (16x256) @ (256x128) ----
  v16bf A2[8];
  #pragma unroll
  for (int kt = 0; kt < 8; ++kt) A2[kt] = afrag_from_lds(h1, 256 / 4, col, kt, klo);
  for (int nt = wave; nt < 8; nt += 4) {
    v8f acc = {};
    #pragma unroll
    for (int kt = 0; kt < 8; ++kt)
      acc = WMMA_BF16(A2[kt], load_frag(pW2, kt, 8, nt, lane), acc);
    float bias = b2[nt * 16 + col];
    #pragma unroll
    for (int v = 0; v < 8; ++v)
      h2[(v + mrow) * 128 + nt * 16 + col] = fmaxf(acc[v] + bias, 0.0f);
  }
  __syncthreads();

  // ---- layer 3: (16x128) @ (128x64); nt == wave (exactly 4 waves) ----
  {
    v16bf A3[4];
    #pragma unroll
    for (int kt = 0; kt < 4; ++kt) A3[kt] = afrag_from_lds(h2, 128 / 4, col, kt, klo);
    int nt = wave;
    v8f acc = {};
    #pragma unroll
    for (int kt = 0; kt < 4; ++kt)
      acc = WMMA_BF16(A3[kt], load_frag(pW3, kt, 4, nt, lane), acc);
    float bias = b3[nt * 16 + col];
    #pragma unroll
    for (int v = 0; v < 8; ++v)
      h3[(v + mrow) * 64 + nt * 16 + col] = fmaxf(acc[v] + bias, 0.0f);
  }
  __syncthreads();

  // ---- final: 64 -> 1 dot + sigmoid ----
  if (tid < 16) {
    float s = bfs[0];
    for (int d2 = 0; d2 < 64; ++d2) s += h3[tid * 64 + d2] * wf[d2];
    out[b0 + tid] = 1.0f / (1.0f + __expf(-s));
  }
}

// ---------------------------------------------------------------------------
extern "C" void kernel_launch(void* const* d_in, const int* in_sizes, int n_in,
                              void* d_out, int out_size, void* d_ws, size_t ws_size,
                              hipStream_t stream) {
  (void)in_sizes; (void)n_in; (void)out_size; (void)ws_size;
  const int*   sx      = (const int*)d_in[0];
  const float* dx      = (const float*)d_in[1];
  const float* emb     = (const float*)d_in[2];
  const float* att_w_W = (const float*)d_in[3];
  const float* att_w_b = (const float*)d_in[4];
  const float* att_d_W = (const float*)d_in[5];
  const float* att_d_b = (const float*)d_in[6];
  const float* bn_g    = (const float*)d_in[7];
  const float* bn_b    = (const float*)d_in[8];
  const float* w1      = (const float*)d_in[9];
  const float* b1      = (const float*)d_in[10];
  const float* w2      = (const float*)d_in[11];
  const float* b2      = (const float*)d_in[12];
  const float* w3      = (const float*)d_in[13];
  const float* b3      = (const float*)d_in[14];
  const float* wf      = (const float*)d_in[15];
  const float* bfs     = (const float*)d_in[16];

  float* out = (float*)d_out;
  float* ws  = (float*)d_ws;
  float* X   = ws;                 // 4096 x 80 (stride-padded 77)
  float* scl = ws + kB * kXS;      // 77
  float* shf = scl + kXS;          // 77
  unsigned short* pWa = (unsigned short*)(shf + kXS);
  unsigned short* pW1 = pWa + kPWa;
  unsigned short* pW2 = pW1 + kPW1;
  unsigned short* pW3 = pW2 + kPW2;

  // pre-pack weights to fragment-native bf16 (cheap; deterministic each call)
  afm_pack<<<(2 * 4 * 32 + 255) / 256, 256, 0, stream>>>(att_w_W, pWa, 64, 2, 64);
  afm_pack<<<(3 * 16 * 32 + 255) / 256, 256, 0, stream>>>(w1, pW1, 77, 3, 256);
  afm_pack<<<(8 * 8 * 32 + 255) / 256, 256, 0, stream>>>(w2, pW2, 256, 8, 128);
  afm_pack<<<(4 * 4 * 32 + 255) / 256, 256, 0, stream>>>(w3, pW3, 128, 4, 64);

  afm_attention<<<kB, 256, 0, stream>>>(sx, dx, emb, pWa, att_w_b,
                                        att_d_W, att_d_b, X);
  afm_bnstats<<<77, 256, 0, stream>>>(X, bn_g, bn_b, scl, shf);
  afm_mlp<<<kB / 16, 128, 0, stream>>>(X, scl, shf, pW1, b1, pW2, b2, pW3, b3,
                                       wf, bfs, out);
}
